// subHRGAT_21406117003314
// MI455X (gfx1250) — matrix operand order repaired and checked by reading, not compiled, not run on hardware
//
#include <hip/hip_runtime.h>
#include <hip/hip_bf16.h>

#define NTILE  50000
#define NPOI   50000
#define NEDGE  200000
#define FOUT   256
#define HEADS  8
#define DHEAD  32
#define NEG_SLOPE 0.2f

typedef __attribute__((ext_vector_type(16))) __bf16 v16bf;
typedef __attribute__((ext_vector_type(8)))  __bf16 v8bf;
typedef __attribute__((ext_vector_type(8)))  float  v8f;

union V16U { v16bf v; v8bf h[2]; };

// ---------------------------------------------------------------- utilities

__global__ void f32_to_bf16_kernel(const float* __restrict__ x, __bf16* __restrict__ y, int n) {
    int i = blockIdx.x * blockDim.x + threadIdx.x;
    if (i < n) y[i] = (__bf16)x[i];
}

// W is (K x N) row-major fp32 -> Wt is (N x K) row-major bf16
__global__ void w_transpose_bf16_kernel(const float* __restrict__ W, __bf16* __restrict__ Wt,
                                        int K, int N) {
    int i = blockIdx.x * blockDim.x + threadIdx.x;
    if (i >= K * N) return;
    int k = i / N, n = i - k * N;
    Wt[(size_t)n * K + k] = (__bf16)W[i];
}

__global__ void fill_f32_kernel(float* __restrict__ p, float v, int n) {
    int i = blockIdx.x * blockDim.x + threadIdx.x;
    if (i < n) p[i] = v;
}

__global__ void scale_relu_kernel(const float* __restrict__ a, float* __restrict__ o,
                                  float s, int n) {
    int i = blockIdx.x * blockDim.x + threadIdx.x;
    if (i >= n) return;
    float v = a[i] * s;
    o[i] = v > 0.0f ? v : 0.0f;
}

// ---------------------------------------------------------------- WMMA GEMM
// C[M,N] = A[M,K](bf16) @ Bt[N,K](bf16, pre-transposed) + bias[N], f32 accumulate.
// One wave per 32x32 C tile (2x2 register blocking, 4 independent accumulators):
// each K-step loads 2 A fragments + 2 B fragments and issues 4 WMMAs, halving
// operand bytes per v_wmma_f32_16x16x32_bf16 vs a 16x16 wave tile and giving the
// XDL 4 independent accumulation chains to pipeline.
__global__ void gemm_bf16_wmma(const __bf16* __restrict__ A, const __bf16* __restrict__ Bt,
                               const float* __restrict__ bias, float* __restrict__ C,
                               int M, int N, int K) {
    int wid  = (blockIdx.x * blockDim.x + threadIdx.x) >> 5;
    int lane = threadIdx.x & 31;
    int ntn  = N >> 5;                 // number of 32-wide column tiles
    int tm   = wid / ntn;
    int tn   = wid - tm * ntn;
    if (tm * 32 >= M) return;          // wave-uniform: EXEC stays all-1s below

    int half = lane >> 4;              // 0: lanes 0-15, 1: lanes 16-31
    int l15  = lane & 15;

    // A fragments: rows tm*32+l15 and tm*32+16+l15 (second clamped; stores guarded)
    int r0 = tm * 32 + l15;
    int r1 = r0 + 16;
    if (r1 >= M) r1 = M - 1;
    const __bf16* arow0 = A + (size_t)r0 * K;
    const __bf16* arow1 = A + (size_t)r1 * K;
    // B fragments: columns tn*32+l15 and tn*32+16+l15 (Bt is N x K, contiguous K)
    const __bf16* brow0 = Bt + (size_t)(tn * 32 + l15) * K;
    const __bf16* brow1 = Bt + (size_t)(tn * 32 + 16 + l15) * K;

    v8f acc00 = {}, acc01 = {}, acc10 = {}, acc11 = {};
    for (int k0 = 0; k0 < K; k0 += 32) {
        V16U a0, a1, b0, b1;
        // 16-bit A layout (ISA 7.12.2): per lane two 8-elem chunks
        a0.h[0] = *(const v8bf*)(arow0 + k0 + (half << 3));
        a0.h[1] = *(const v8bf*)(arow0 + k0 + 16 + (half << 3));
        a1.h[0] = *(const v8bf*)(arow1 + k0 + (half << 3));
        a1.h[1] = *(const v8bf*)(arow1 + k0 + 16 + (half << 3));
        // 16-bit B layout: lane holds one column, same two 8-elem K chunks as A
        b0.h[0] = *(const v8bf*)(brow0 + k0 + (half << 3));
        b0.h[1] = *(const v8bf*)(brow0 + k0 + 16 + (half << 3));
        b1.h[0] = *(const v8bf*)(brow1 + k0 + (half << 3));
        b1.h[1] = *(const v8bf*)(brow1 + k0 + 16 + (half << 3));
        if (k0 + 32 < K) {
            __builtin_prefetch(arow0 + k0 + 32, 0, 3);
            __builtin_prefetch(arow1 + k0 + 32, 0, 3);
        }
        acc00 = __builtin_amdgcn_wmma_f32_16x16x32_bf16(false, a0.v, false, b0.v, (short)0, acc00, false, false);
        acc01 = __builtin_amdgcn_wmma_f32_16x16x32_bf16(false, a0.v, false, b1.v, (short)0, acc01, false, false);
        acc10 = __builtin_amdgcn_wmma_f32_16x16x32_bf16(false, a1.v, false, b0.v, (short)0, acc10, false, false);
        acc11 = __builtin_amdgcn_wmma_f32_16x16x32_bf16(false, a1.v, false, b1.v, (short)0, acc11, false, false);
    }

    // C layout: VGPR r -> row = rowbase + half*8 + r, col = colbase + l15
    int c0 = tn * 32 + l15;
    int c1 = c0 + 16;
    float bias0 = bias[c0], bias1 = bias[c1];
    int row0 = tm * 32 + half * 8;      // rows row0..row0+7 always < M (M % 16 == 0)
    int row1 = row0 + 16;               // may spill past M -> guarded
#pragma unroll
    for (int r = 0; r < 8; ++r) {
        size_t rw0 = (size_t)(row0 + r) * N;
        C[rw0 + c0] = acc00[r] + bias0;
        C[rw0 + c1] = acc01[r] + bias1;
        int rr1 = row1 + r;
        if (rr1 < M) {
            size_t rw1 = (size_t)rr1 * N;
            C[rw1 + c0] = acc10[r] + bias0;
            C[rw1 + c1] = acc11[r] + bias1;
        }
    }
}

// ---------------------------------------------------------------- edge phase

__device__ __forceinline__ void atomicMaxF(float* addr, float val) {
    if (val >= 0.0f) atomicMax((int*)addr, __float_as_int(val));
    else             atomicMin((unsigned int*)addr, __float_as_uint(val));
}

// warp-per-edge: logits[e][h] = sum_d attn[h][d]*leakyrelu(fs[src][h,d]+fd[dst][h,d]);
// also atomicMax into mx[dst][h].
__global__ void edge_logits_max(const float* __restrict__ fs, const float* __restrict__ fd,
                                const int* __restrict__ src, const int* __restrict__ dst,
                                const float* __restrict__ attn,
                                float* __restrict__ logits, float* __restrict__ mx, int E) {
    int gid  = blockIdx.x * blockDim.x + threadIdx.x;
    int e    = gid >> 5;
    int lane = gid & 31;
    if (e >= E) return;
    const float* pfs = fs + (size_t)src[e] * FOUT;
    const float* pfd = fd + (size_t)dst[e] * FOUT;
    int dbase = dst[e] * HEADS;
#pragma unroll
    for (int h = 0; h < HEADS; ++h) {
        float v = pfs[h * DHEAD + lane] + pfd[h * DHEAD + lane];
        v = v > 0.0f ? v : v * NEG_SLOPE;
        v *= attn[h * DHEAD + lane];
#pragma unroll
        for (int off = 16; off; off >>= 1) v += __shfl_xor(v, off, 32);
        if (lane == 0) {
            logits[(size_t)e * HEADS + h] = v;
            atomicMaxF(&mx[dbase + h], v);
        }
    }
}

// thread per (edge,head): ex = exp(logit - mx[dst]); den[dst] += ex
__global__ void edge_exp_den(const float* __restrict__ logits, const int* __restrict__ dst,
                             const float* __restrict__ mx, float* __restrict__ ex,
                             float* __restrict__ den, int E8) {
    int i = blockIdx.x * blockDim.x + threadIdx.x;
    if (i >= E8) return;
    int e = i >> 3, h = i & 7;
    int d = dst[e];
    float v = __expf(logits[i] - mx[d * HEADS + h]);
    ex[i] = v;
    unsafeAtomicAdd(&den[d * HEADS + h], v);
}

// warp-per-edge: acc[dst] += fs[src] * (ex/den[dst]) per head
__global__ void edge_aggregate(const float* __restrict__ fs, const int* __restrict__ src,
                               const int* __restrict__ dst, const float* __restrict__ ex,
                               const float* __restrict__ den, float* __restrict__ acc, int E) {
    int gid  = blockIdx.x * blockDim.x + threadIdx.x;
    int e    = gid >> 5;
    int lane = gid & 31;
    if (e >= E) return;
    const float* pfs = fs + (size_t)src[e] * FOUT;
    int d = dst[e];
    float* pacc = acc + (size_t)d * FOUT;
#pragma unroll
    for (int h = 0; h < HEADS; ++h) {
        float alpha = ex[(size_t)e * HEADS + h] / den[d * HEADS + h];
        unsafeAtomicAdd(&pacc[h * DHEAD + lane], pfs[h * DHEAD + lane] * alpha);
    }
}

// ---------------------------------------------------------------- host driver

extern "C" void kernel_launch(void* const* d_in, const int* in_sizes, int n_in,
                              void* d_out, int out_size, void* d_ws, size_t ws_size,
                              hipStream_t stream) {
    const float* x_tile = (const float*)d_in[0];
    const float* x_poi  = (const float*)d_in[1];
    const int* src_e[3] = { (const int*)d_in[2], (const int*)d_in[4], (const int*)d_in[6] };
    const int* dst_e[3] = { (const int*)d_in[3], (const int*)d_in[5], (const int*)d_in[7] };
    const float* Wsrc[3] = { (const float*)d_in[8],  (const float*)d_in[13], (const float*)d_in[18] };
    const float* bsrc[3] = { (const float*)d_in[9],  (const float*)d_in[14], (const float*)d_in[19] };
    const float* Wdst[3] = { (const float*)d_in[10], (const float*)d_in[15], (const float*)d_in[20] };
    const float* bdst[3] = { (const float*)d_in[11], (const float*)d_in[16], (const float*)d_in[21] };
    const float* attn[3] = { (const float*)d_in[12], (const float*)d_in[17], (const float*)d_in[22] };

    // workspace carve-up (256B aligned)
    char* wsb = (char*)d_ws;
    size_t off = 0;
    auto take = [&](size_t bytes) -> char* {
        char* q = wsb + off;
        off += (bytes + 255) & ~(size_t)255;
        return q;
    };
    float*  acc_t  = (float*) take((size_t)NTILE * FOUT * 4);   // doubles as ht between layers
    float*  acc_p  = (float*) take((size_t)NPOI  * FOUT * 4);   // doubles as hp
    __bf16* ht_bf  = (__bf16*)take((size_t)NTILE * FOUT * 2);
    __bf16* hp_bf  = (__bf16*)take((size_t)NPOI  * FOUT * 2);
    float*  fsb    = (float*) take((size_t)NTILE * FOUT * 4);
    float*  fdb    = (float*) take((size_t)NTILE * FOUT * 4);
    float*  logits = (float*) take((size_t)NEDGE * HEADS * 4);
    float*  exb    = (float*) take((size_t)NEDGE * HEADS * 4);
    float*  mx     = (float*) take((size_t)NTILE * HEADS * 4);
    float*  den    = (float*) take((size_t)NTILE * HEADS * 4);
    __bf16* WtS    = (__bf16*)take((size_t)256 * 256 * 2);
    __bf16* WtD    = (__bf16*)take((size_t)256 * 256 * 2);

    const __bf16* sfeat[3] = { ht_bf, hp_bf, ht_bf };
    const __bf16* dfeat[3] = { ht_bf, ht_bf, hp_bf };
    const int Ns[3] = { NTILE, NPOI, NTILE };
    const int Nd[3] = { NTILE, NTILE, NPOI };
    float* accs[3] = { acc_t, acc_t, acc_p };

    const int B = 256;
    const float NEGINF = -3.402823466e38f;

    for (int l = 0; l < 3; ++l) {
        int fin = (l == 0) ? 128 : 256;
        const float* ht_in = (l == 0) ? x_tile : acc_t;
        const float* hp_in = (l == 0) ? x_poi  : acc_p;

        // convert current activations to bf16 (must precede zeroing acc)
        f32_to_bf16_kernel<<<((size_t)NTILE * fin + B - 1) / B, B, 0, stream>>>(ht_in, ht_bf, NTILE * fin);
        f32_to_bf16_kernel<<<((size_t)NPOI  * fin + B - 1) / B, B, 0, stream>>>(hp_in, hp_bf, NPOI * fin);
        fill_f32_kernel<<<((size_t)NTILE * FOUT + B - 1) / B, B, 0, stream>>>(acc_t, 0.0f, NTILE * FOUT);
        fill_f32_kernel<<<((size_t)NPOI  * FOUT + B - 1) / B, B, 0, stream>>>(acc_p, 0.0f, NPOI * FOUT);

        for (int r = 0; r < 3; ++r) {
            size_t woff = (size_t)r * fin * FOUT;
            w_transpose_bf16_kernel<<<(fin * FOUT + B - 1) / B, B, 0, stream>>>(Wsrc[l] + woff, WtS, fin, FOUT);
            w_transpose_bf16_kernel<<<(fin * FOUT + B - 1) / B, B, 0, stream>>>(Wdst[l] + woff, WtD, fin, FOUT);

            // one wave per 32x32 output tile
            int waves_s = ((Ns[r] + 31) / 32) * (FOUT / 32);
            int waves_d = ((Nd[r] + 31) / 32) * (FOUT / 32);
            gemm_bf16_wmma<<<(waves_s + 7) / 8, B, 0, stream>>>(sfeat[r], WtS, bsrc[l] + r * FOUT, fsb, Ns[r], FOUT, fin);
            gemm_bf16_wmma<<<(waves_d + 7) / 8, B, 0, stream>>>(dfeat[r], WtD, bdst[l] + r * FOUT, fdb, Nd[r], FOUT, fin);

            fill_f32_kernel<<<(Nd[r] * HEADS + B - 1) / B, B, 0, stream>>>(mx, NEGINF, Nd[r] * HEADS);
            fill_f32_kernel<<<(Nd[r] * HEADS + B - 1) / B, B, 0, stream>>>(den, 0.0f, Nd[r] * HEADS);

            edge_logits_max<<<((size_t)NEDGE * 32 + B - 1) / B, B, 0, stream>>>(
                fsb, fdb, src_e[r], dst_e[r], attn[l] + (size_t)r * HEADS * DHEAD, logits, mx, NEDGE);
            edge_exp_den<<<((size_t)NEDGE * HEADS + B - 1) / B, B, 0, stream>>>(
                logits, dst_e[r], mx, exb, den, NEDGE * HEADS);
            edge_aggregate<<<((size_t)NEDGE * 32 + B - 1) / B, B, 0, stream>>>(
                fsb, src_e[r], dst_e[r], exb, den, accs[r], NEDGE);
        }

        float tscale = (l == 1) ? 0.5f : 1.0f;
        float* out_t = (l == 2) ? (float*)d_out : acc_t;
        float* out_p = (l == 2) ? ((float*)d_out + (size_t)NTILE * FOUT) : acc_p;
        scale_relu_kernel<<<((size_t)NTILE * FOUT + B - 1) / B, B, 0, stream>>>(acc_t, out_t, tscale, NTILE * FOUT);
        scale_relu_kernel<<<((size_t)NPOI  * FOUT + B - 1) / B, B, 0, stream>>>(acc_p, out_p, 1.0f, NPOI * FOUT);
    }
}
